// NTM_77945066487891
// MI455X (gfx1250) — matrix operand order repaired and checked by reading, not compiled, not run on hardware
//
#include <hip/hip_runtime.h>
#include <hip/hip_bf16.h>
#include <math.h>

// ---------------------------------------------------------------------------
// NTM forward step for MI455X (gfx1250, wave32, WMMA + async global->LDS).
// Sizes fixed by the reference: B=256, N=32768, M=64, R=4, U=256, Din=512.
// ---------------------------------------------------------------------------

#define B_    256
#define N_    32768
#define M_    64
#define R_    4
#define U_    256
#define DIN_  512
#define KCAT_ 1024      // Din + R*M + U  (stacked LSTM K)
#define G4_   1024      // 4*U
#define RB_   1024      // R*B
#define OUTW_ 512       // U + R*M

typedef __attribute__((ext_vector_type(16))) __bf16 v16bf;
typedef __attribute__((ext_vector_type(8)))  float  v8f;

union FragBF { v16bf v; uint4 q[2]; __bf16 h[16]; };

// ---- WMMA helpers ----------------------------------------------------------
// A-matrix 16x32 bf16 (ISA 7.12.2): lane<16 -> row m=lane, K in [0,8)U[16,24);
// lane>=16 -> row m=lane-16, K in [8,16)U[24,32). Two 16B contiguous loads.
__device__ __forceinline__ v16bf load_a_frag(const __bf16* A, long lda,
                                             int row0, int k0) {
  int lane = threadIdx.x & 31;
  int m  = lane & 15;
  int kb = (lane < 16) ? 0 : 8;
  const __bf16* p = A + (long)(row0 + m) * lda + k0 + kb;
  FragBF f;
  f.q[0] = *(const uint4*)(p);        // K group 1 (8 bf16)
  f.q[1] = *(const uint4*)(p + 16);   // K group 2 (8 bf16)
  return f.v;
}

// B-matrix 32x16 bf16: lane<16 -> col n=lane, K in [0,8)U[16,24); lane>=16 -> col
// n=lane-16, K in [8,16)U[24,32). BT is stored column-major (col, then K contiguous).
__device__ __forceinline__ v16bf load_b_frag(const __bf16* BT, long ldb,
                                             int col0, int k0) {
  int lane = threadIdx.x & 31;
  int n  = lane & 15;
  int kb = (lane < 16) ? 0 : 8;
  const __bf16* p = BT + (long)(col0 + n) * ldb + k0 + kb;
  FragBF f;
  f.q[0] = *(const uint4*)(p);
  f.q[1] = *(const uint4*)(p + 16);
  return f.v;
}

__device__ __forceinline__ v8f wmma_bf16(v16bf a, v16bf b, v8f c) {
  return __builtin_amdgcn_wmma_f32_16x16x32_bf16(
      /*neg_a=*/false, a, /*neg_b=*/false, b,
      /*c_mod=*/(short)0, c, /*reuse_a=*/false, /*reuse_b=*/false);
}

__device__ __forceinline__ void zero_acc(v8f* acc, int n) {
  for (int cc = 0; cc < n; ++cc)
    for (int e = 0; e < 8; ++e) acc[cc][e] = 0.0f;
}

__device__ __forceinline__ float sigmoidf_(float x) {
  return 1.0f / (1.0f + __expf(-x));
}

// ---- async global -> LDS (gfx1250, ASYNCcnt) -------------------------------
// the LDS address comes from the VDST VGPR; generic LDS pointers carry the LDS offset in
// their low 32 bits (ISA 10.2 aperture rules), so truncation gives VDST.
__device__ __forceinline__ void async_copy_b128(void* lds_ptr, const void* gptr) {
  unsigned int  loff = (unsigned int)(uintptr_t)lds_ptr;
  unsigned long long ga = (unsigned long long)(uintptr_t)gptr;
  asm volatile("global_load_async_to_lds_b128 %0, %1, off"
               :: "v"(loff), "v"(ga) : "memory");
}
__device__ __forceinline__ void wait_async_le4() {
  asm volatile("s_wait_asynccnt 0x4" ::: "memory");
}
__device__ __forceinline__ void wait_async_0() {
  asm volatile("s_wait_asynccnt 0x0" ::: "memory");
}

// ---------------------------------------------------------------------------
// Prep: stacked controller input [x | read_prev(b-major) | h] as bf16 [B,1024]
// ---------------------------------------------------------------------------
__global__ void k_prep_catA(const float* __restrict__ x,
                            const float* __restrict__ read_prev,
                            const float* __restrict__ h,
                            __bf16* __restrict__ catA) {
  int t = blockIdx.x * blockDim.x + threadIdx.x;
  if (t >= B_ * KCAT_) return;
  int b = t >> 10, k = t & 1023;
  float v;
  if (k < DIN_) {
    v = x[(long)b * DIN_ + k];
  } else if (k < DIN_ + R_ * M_) {
    int j = k - DIN_; int r = j >> 6, m = j & 63;
    v = read_prev[((long)(r * B_ + b)) * M_ + m];   // [R,B,M]
  } else {
    v = h[(long)b * U_ + (k - DIN_ - R_ * M_)];
  }
  catA[t] = (__bf16)v;
}

// Stacked weight, transposed: WcatT[j][k] = (k<768 ? W[k][j] : Uw[k-768][j])
__global__ void k_prep_wcatT(const float* __restrict__ W,
                             const float* __restrict__ Uw,
                             __bf16* __restrict__ WT) {
  int t = blockIdx.x * blockDim.x + threadIdx.x;
  if (t >= KCAT_ * G4_) return;
  int j = t >> 10, k = t & 1023;
  float v = (k < DIN_ + R_ * M_) ? W[(long)k * G4_ + j]
                                 : Uw[(long)(k - DIN_ - R_ * M_) * G4_ + j];
  WT[(long)j * KCAT_ + k] = (__bf16)v;
}

// ---------------------------------------------------------------------------
// LSTM gates GEMM: C[256,1024] = catA[256,1024K] * Wcat ; bf16 WMMA, f32 acc.
// K loop unrolled x2 with alternating fragment sets (no register rotation).
// ---------------------------------------------------------------------------
__global__ void k_gemm_gates(const __bf16* __restrict__ A,
                             const __bf16* __restrict__ BT,
                             float* __restrict__ C) {
  int wave = blockIdx.x * (blockDim.x >> 5) + (threadIdx.x >> 5);
  int rowTile = wave >> 4;          // 16 row tiles of 16
  int colGrp  = wave & 15;          // 16 col groups of 64
  int row0 = rowTile * 16, col0 = colGrp * 64;
  v8f acc[4]; zero_acc(acc, 4);

  v16bf aA = load_a_frag(A, KCAT_, row0, 0);
  v16bf bA[4];
#pragma unroll
  for (int cc = 0; cc < 4; ++cc) bA[cc] = load_b_frag(BT, KCAT_, col0 + cc * 16, 0);

  for (int k0 = 0; k0 < KCAT_; k0 += 64) {
    v16bf aB = load_a_frag(A, KCAT_, row0, k0 + 32);
    v16bf bB[4];
#pragma unroll
    for (int cc = 0; cc < 4; ++cc) bB[cc] = load_b_frag(BT, KCAT_, col0 + cc * 16, k0 + 32);
#pragma unroll
    for (int cc = 0; cc < 4; ++cc) acc[cc] = wmma_bf16(aA, bA[cc], acc[cc]);
    int k2 = k0 + 64;
    if (k2 < KCAT_) {
      aA = load_a_frag(A, KCAT_, row0, k2);
#pragma unroll
      for (int cc = 0; cc < 4; ++cc) bA[cc] = load_b_frag(BT, KCAT_, col0 + cc * 16, k2);
    }
#pragma unroll
    for (int cc = 0; cc < 4; ++cc) acc[cc] = wmma_bf16(aB, bB[cc], acc[cc]);
  }

  int lane = threadIdx.x & 31;
  int nl = lane & 15;
  int mb = row0 + ((lane < 16) ? 0 : 8);
#pragma unroll
  for (int cc = 0; cc < 4; ++cc) {
    int col = col0 + cc * 16 + nl;
#pragma unroll
    for (int i = 0; i < 8; ++i)
      C[(long)(mb + i) * G4_ + col] = acc[cc][i];
  }
}

// LSTM elementwise: h_new, and write out[:, :256]
__global__ void k_lstm(const float* __restrict__ gates,
                       const float* __restrict__ bvec,
                       const float* __restrict__ c,
                       float* __restrict__ h_new,
                       float* __restrict__ out) {
  int t = blockIdx.x * blockDim.x + threadIdx.x;
  if (t >= B_ * U_) return;
  int b = t >> 8, u = t & 255;
  const float* g = gates + (long)b * G4_;
  float gi = g[u]           + bvec[u];
  float gf = g[U_ + u]      + bvec[U_ + u];
  float gg = g[2 * U_ + u]  + bvec[2 * U_ + u];
  float go = g[3 * U_ + u]  + bvec[3 * U_ + u];
  float cn = sigmoidf_(gf) * c[t] + sigmoidf_(gi) * tanhf(gg);
  float hn = sigmoidf_(go) * tanhf(cn);
  h_new[t] = hn;
  out[(long)b * OUTW_ + u] = hn;
}

// ---------------------------------------------------------------------------
// kt = tanh(h_new @ Wk[r] + bk): tiny GEMM, plain VALU; writes kt and ktT bf16.
// ---------------------------------------------------------------------------
__global__ void k_kt(const float* __restrict__ h_new,
                     const float* __restrict__ Wk,
                     const float* __restrict__ bk,
                     __bf16* __restrict__ kt,
                     __bf16* __restrict__ ktT) {
  int t = blockIdx.x * blockDim.x + threadIdx.x;
  if (t >= R_ * B_ * M_) return;
  int r = t >> 14; int rem = t & 16383; int b = rem >> 6, m = rem & 63;
  const float* hr = h_new + (long)b * U_;
  const float* wc = Wk + (long)r * U_ * M_ + m;   // stride M_ over u
  float s = bk[r * M_ + m];
  for (int u = 0; u < U_; ++u) s += hr[u] * wc[(long)u * M_];
  float v = tanhf(s);
  kt[((long)(r * B_ + b)) * M_ + m]  = (__bf16)v;
  ktT[((long)(r * M_ + m)) * B_ + b] = (__bf16)v;
}

__global__ void k_knorm(const __bf16* __restrict__ kt,
                        float* __restrict__ knorm) {
  int rb = blockIdx.x * blockDim.x + threadIdx.x;
  if (rb >= RB_) return;
  float s = 0.0f;
  for (int m = 0; m < M_; ++m) {
    float v = (float)kt[(long)rb * M_ + m];
    s += v * v;
  }
  knorm[rb] = sqrtf(s);
}

__global__ void k_alpha(const float* __restrict__ h_new,
                        const float* __restrict__ Wa,
                        const float* __restrict__ ba,
                        float* __restrict__ alpha) {
  int t = blockIdx.x * blockDim.x + threadIdx.x;
  if (t >= B_ * R_) return;
  int b = t >> 2, r = t & 3;
  float s = ba[r];
  for (int u = 0; u < U_; ++u) s += h_new[(long)b * U_ + u] * Wa[u * R_ + r];
  alpha[t] = sigmoidf_(tanhf(s));
}

// ---------------------------------------------------------------------------
// argmin over N per batch row (first-occurrence tie-break), then erase flags.
// ---------------------------------------------------------------------------
__global__ void k_argmin(const float* __restrict__ wu, int* __restrict__ least) {
  __shared__ float sv[256];
  __shared__ int   si[256];
  int b = blockIdx.x, t = threadIdx.x;
  const float* row = wu + (long)b * N_;
  float mv = row[t]; int mi = t;
  for (int n = t + 256; n < N_; n += 256) {
    float v = row[n];
    if (v < mv) { mv = v; mi = n; }
  }
  sv[t] = mv; si[t] = mi;
  __syncthreads();
  for (int s = 128; s > 0; s >>= 1) {
    if (t < s) {
      float v2 = sv[t + s]; int i2 = si[t + s];
      if (v2 < sv[t] || (v2 == sv[t] && i2 < si[t])) { sv[t] = v2; si[t] = i2; }
    }
    __syncthreads();
  }
  if (t == 0) least[b] = si[0];
}

__global__ void k_erase_init(float* __restrict__ erase) {
  int n = blockIdx.x * blockDim.x + threadIdx.x;
  if (n < N_) erase[n] = 1.0f;
}
__global__ void k_erase_scatter(const int* __restrict__ least,
                                float* __restrict__ erase) {
  int b = threadIdx.x;
  if (b < B_) erase[least[b]] = 0.0f;   // idempotent writes
}

// A -> bf16 copy + row norms. One wave (32 lanes) per row of 64.
__global__ void k_aprep(const float* __restrict__ A,
                        __bf16* __restrict__ Abf,
                        float* __restrict__ anorm) {
  int n = blockIdx.x, l = threadIdx.x;
  float v0 = A[(long)n * M_ + l];
  float v1 = A[(long)n * M_ + 32 + l];
  Abf[(long)n * M_ + l]      = (__bf16)v0;
  Abf[(long)n * M_ + 32 + l] = (__bf16)v1;
  float s = v0 * v0 + v1 * v1;
  for (int o = 16; o > 0; o >>= 1) s += __shfl_down(s, o, 32);
  if (l == 0) anorm[n] = sqrtf(s);
}

// ---------------------------------------------------------------------------
// Content addressing GEMM + cosine scale:
//   Ksim[rb,n] = (kt[rb,:]·A[n,:]) / (knorm[rb]*anorm[n] + 1e-16)
// M=1024 rows, N=32768 cols, K=64 (both K-steps' fragments issued up front).
// ---------------------------------------------------------------------------
__global__ void k_gemm_ksim(const __bf16* __restrict__ ktbf,
                            const __bf16* __restrict__ Abf,
                            const float* __restrict__ knorm,
                            const float* __restrict__ anorm,
                            float* __restrict__ Ksim) {
  int wave = blockIdx.x * (blockDim.x >> 5) + (threadIdx.x >> 5);
  int rowTile = wave >> 9;          // 64 row tiles
  int colGrp  = wave & 511;         // 512 col groups of 64
  int row0 = rowTile * 16, col0 = colGrp * 64;
  v8f acc[4]; zero_acc(acc, 4);
  v16bf a0 = load_a_frag(ktbf, M_, row0, 0);
  v16bf a1 = load_a_frag(ktbf, M_, row0, 32);
  v16bf b0[4], b1[4];
#pragma unroll
  for (int cc = 0; cc < 4; ++cc) {
    b0[cc] = load_b_frag(Abf, M_, col0 + cc * 16, 0);
    b1[cc] = load_b_frag(Abf, M_, col0 + cc * 16, 32);
  }
#pragma unroll
  for (int cc = 0; cc < 4; ++cc) acc[cc] = wmma_bf16(a0, b0[cc], acc[cc]);
#pragma unroll
  for (int cc = 0; cc < 4; ++cc) acc[cc] = wmma_bf16(a1, b1[cc], acc[cc]);

  int lane = threadIdx.x & 31;
  int nl = lane & 15;
  int mb = row0 + ((lane < 16) ? 0 : 8);
  float kn8[8];
#pragma unroll
  for (int i = 0; i < 8; ++i) kn8[i] = knorm[mb + i];
#pragma unroll
  for (int cc = 0; cc < 4; ++cc) {
    int col = col0 + cc * 16 + nl;
    float an = anorm[col];
#pragma unroll
    for (int i = 0; i < 8; ++i)
      Ksim[(long)(mb + i) * N_ + col] = acc[cc][i] / (kn8[i] * an + 1e-16f);
  }
}

// Row softmax over N=32768 (online two-pass), writes wr as bf16.
__global__ void k_softmax(const float* __restrict__ Ksim,
                          __bf16* __restrict__ wr) {
  int rb = blockIdx.x, t = threadIdx.x;
  const float* row = Ksim + (long)rb * N_;
  float m = -3.4e38f, s = 0.0f;
  for (int n = t; n < N_; n += 256) {
    float x = row[n];
    if (x > m) { s = s * __expf(m - x) + 1.0f; m = x; }
    else       { s += __expf(x - m); }
  }
  for (int o = 16; o > 0; o >>= 1) {
    float m2 = __shfl_down(m, o, 32), s2 = __shfl_down(s, o, 32);
    float MM = fmaxf(m, m2);
    s = s * __expf(m - MM) + s2 * __expf(m2 - MM);
    m = MM;
  }
  __shared__ float sm[8], ss[8], gmv, gsv;
  int w = t >> 5, l = t & 31;
  if (l == 0) { sm[w] = m; ss[w] = s; }
  __syncthreads();
  if (t == 0) {
    float MM = sm[0], SS = ss[0];
    for (int i = 1; i < 8; ++i) {
      float Mi = sm[i], M2 = fmaxf(MM, Mi);
      SS = SS * __expf(MM - M2) + ss[i] * __expf(Mi - M2);
      MM = M2;
    }
    gmv = MM; gsv = 1.0f / SS;
  }
  __syncthreads();
  float GM = gmv, GSI = gsv;
  __bf16* wrow = wr + (long)rb * N_;
  for (int n = t; n < N_; n += 256)
    wrow[n] = (__bf16)(__expf(row[n] - GM) * GSI);
}

// ---------------------------------------------------------------------------
// Memory update (one r per launch, sequential):
//   A_new = tanh(prev + ww[r]^T @ kt[r]),  prev = first ? A*erase : A_new
// Block = 128 thr (4 waves), 64 rows (n) x 64 cols (m), K=B=256.
// wr_prev chunks are brought in with ASYNC global->LDS copies (double
// buffered, overlapped with transform+WMMA of the current chunk), then the
// ww tile (alpha*wr_prev + (1-alpha)*onehot) is written transposed as bf16 so
// A-fragments are contiguous 16B LDS reads.
// ---------------------------------------------------------------------------
__global__ void k_update(const float* __restrict__ wr_prev_r,
                         const float* __restrict__ alpha,
                         const int* __restrict__ least,
                         const float* __restrict__ erase,
                         const float* __restrict__ A,
                         const __bf16* __restrict__ ktT_r,
                         float* __restrict__ Anew,
                         __bf16* __restrict__ AnewT,
                         int r, int first, int last) {
  __shared__ __align__(16) float  fbuf[2][32 * 64];   // [k (b)][n_local] f32
  __shared__ __align__(16) __bf16 tile[64 * 40];      // [n_local][k] bf16, pad 40
  int n0 = blockIdx.x * 64;
  int t = threadIdx.x;
  int wave = t >> 5, lane = t & 31;
  v8f acc[4]; zero_acc(acc, 4);

  // each thread copies 4 x 16B segments per 32x64 chunk (512 segments total)
  int seg0 = t * 4;

  // issue chunk 0
  {
    const float* base = wr_prev_r + (long)0 * N_ + n0;
#pragma unroll
    for (int i = 0; i < 4; ++i) {
      int s = seg0 + i;
      int kk = s >> 4, part = s & 15;
      async_copy_b128(&fbuf[0][kk * 64 + part * 4],
                      base + (long)kk * N_ + part * 4);
    }
  }

  for (int kc = 0; kc < 8; ++kc) {
    int k0 = kc * 32;
    if (kc < 7) {   // prefetch next chunk into the other buffer
      const float* base = wr_prev_r + (long)(k0 + 32) * N_ + n0;
#pragma unroll
      for (int i = 0; i < 4; ++i) {
        int s = seg0 + i;
        int kk = s >> 4, part = s & 15;
        async_copy_b128(&fbuf[(kc + 1) & 1][kk * 64 + part * 4],
                        base + (long)kk * N_ + part * 4);
      }
      wait_async_le4();   // current chunk's 4 copies done (in-order)
    } else {
      wait_async_0();
    }
    __syncthreads();      // fbuf[kc&1] visible; prior tile reads finished

    // transform: ww = alpha*wr_prev + (1-alpha)*onehot(least), transposed bf16
    {
      int kk = t >> 2, jb = (t & 3) * 16;
      int b = k0 + kk;
      float al = alpha[b * R_ + r];
      float om = 1.0f - al;
      int lst = least[b];
      const float* src = &fbuf[kc & 1][kk * 64];
      for (int j = jb; j < jb + 16; ++j) {
        int n = n0 + j;
        float v = al * src[j] + ((lst == n) ? om : 0.0f);
        tile[j * 40 + kk] = (__bf16)v;
      }
    }
    __syncthreads();

    // A-fragment from LDS (rows = n), B-fragment = ktT (cols = m, K = b)
    int mrow = wave * 16 + (lane & 15);
    int kb = (lane < 16) ? 0 : 8;
    const __bf16* p = &tile[mrow * 40 + kb];
    FragBF fa;
    fa.q[0] = *(const uint4*)(p);
    fa.q[1] = *(const uint4*)(p + 16);
#pragma unroll
    for (int cc = 0; cc < 4; ++cc) {
      v16bf bfr = load_b_frag(ktT_r, B_, cc * 16, k0);
      acc[cc] = wmma_bf16(fa.v, bfr, acc[cc]);
    }
  }

  int nbase = n0 + wave * 16 + ((lane < 16) ? 0 : 8);
  int ml = lane & 15;
#pragma unroll
  for (int cc = 0; cc < 4; ++cc) {
    int mcol = cc * 16 + ml;
#pragma unroll
    for (int i = 0; i < 8; ++i) {
      int nrow = nbase + i;
      float prev = first ? A[(long)nrow * M_ + mcol] * erase[nrow]
                         : Anew[(long)nrow * M_ + mcol];
      float nv = tanhf(prev + acc[cc][i]);
      Anew[(long)nrow * M_ + mcol] = nv;
      if (last) AnewT[(long)mcol * N_ + nrow] = (__bf16)nv;
    }
  }
}

// ---------------------------------------------------------------------------
// reads GEMM (split-K=16 + f32 atomics): reads[rb,m] = sum_n wr[rb,n]*Anew[n,m]
// M=1024 rows, N=64 cols, K=32768. K loop unrolled x2, alternating frag sets.
// ---------------------------------------------------------------------------
__global__ void k_zero_reads(float* __restrict__ reads) {
  int t = blockIdx.x * blockDim.x + threadIdx.x;
  if (t < RB_ * M_) reads[t] = 0.0f;
}

__global__ void k_gemm_reads(const __bf16* __restrict__ wrbf,
                             const __bf16* __restrict__ AnewT,
                             float* __restrict__ reads) {
  int flat = blockIdx.x * (blockDim.x >> 5) + (threadIdx.x >> 5); // [0,1024)
  int rowTile = flat >> 4, split = flat & 15;
  int row0 = rowTile * 16;
  int kbeg = split * (N_ / 16), kend = kbeg + (N_ / 16);
  v8f acc[4]; zero_acc(acc, 4);

  v16bf aA = load_a_frag(wrbf, N_, row0, kbeg);
  v16bf bA[4];
#pragma unroll
  for (int cc = 0; cc < 4; ++cc) bA[cc] = load_b_frag(AnewT, N_, cc * 16, kbeg);

  for (int k0 = kbeg; k0 < kend; k0 += 64) {
    v16bf aB = load_a_frag(wrbf, N_, row0, k0 + 32);
    v16bf bB[4];
#pragma unroll
    for (int cc = 0; cc < 4; ++cc) bB[cc] = load_b_frag(AnewT, N_, cc * 16, k0 + 32);
#pragma unroll
    for (int cc = 0; cc < 4; ++cc) acc[cc] = wmma_bf16(aA, bA[cc], acc[cc]);
    int k2 = k0 + 64;
    if (k2 < kend) {
      aA = load_a_frag(wrbf, N_, row0, k2);
#pragma unroll
      for (int cc = 0; cc < 4; ++cc) bA[cc] = load_b_frag(AnewT, N_, cc * 16, k2);
    }
#pragma unroll
    for (int cc = 0; cc < 4; ++cc) acc[cc] = wmma_bf16(aB, bB[cc], acc[cc]);
  }

  int lane = threadIdx.x & 31;
  int nl = lane & 15;
  int mb = row0 + ((lane < 16) ? 0 : 8);
#pragma unroll
  for (int cc = 0; cc < 4; ++cc) {
    int col = cc * 16 + nl;
#pragma unroll
    for (int i = 0; i < 8; ++i)
      atomicAdd(&reads[(long)(mb + i) * M_ + col], acc[cc][i]);
  }
}

// out[b, 256 + r*64 + m] = reads[r*B+b, m]
__global__ void k_out(const float* __restrict__ reads, float* __restrict__ out) {
  int t = blockIdx.x * blockDim.x + threadIdx.x;
  if (t >= B_ * R_ * M_) return;
  int b = t >> 8, j = t & 255;
  int r = j >> 6, m = j & 63;
  out[(long)b * OUTW_ + U_ + j] = reads[((long)(r * B_ + b)) * M_ + m];
}

// ---------------------------------------------------------------------------
extern "C" void kernel_launch(void* const* d_in, const int* in_sizes, int n_in,
                              void* d_out, int out_size, void* d_ws, size_t ws_size,
                              hipStream_t stream) {
  const float* x         = (const float*)d_in[0];
  const float* A         = (const float*)d_in[1];
  const float* wr_prev   = (const float*)d_in[2];
  const float* wu_prev   = (const float*)d_in[3];
  const float* read_prev = (const float*)d_in[4];
  const float* h         = (const float*)d_in[5];
  const float* c         = (const float*)d_in[6];
  const float* W         = (const float*)d_in[7];
  const float* Uw        = (const float*)d_in[8];
  const float* bvec      = (const float*)d_in[9];
  const float* Wk        = (const float*)d_in[10];
  const float* bk        = (const float*)d_in[11];
  const float* Wa        = (const float*)d_in[12];
  const float* ba        = (const float*)d_in[13];
  float* out = (float*)d_out;

  char* w = (char*)d_ws;
  size_t off = 0;
  auto carve = [&](size_t bytes) -> char* {
    char* p = w + off;
    off = (off + bytes + 255) & ~(size_t)255;
    return p;
  };
  __bf16* catA   = (__bf16*)carve((size_t)B_ * KCAT_ * 2);
  __bf16* wcatT  = (__bf16*)carve((size_t)G4_ * KCAT_ * 2);
  float*  gates  = (float*) carve((size_t)B_ * G4_ * 4);
  float*  h_new  = (float*) carve((size_t)B_ * U_ * 4);
  __bf16* ktbf   = (__bf16*)carve((size_t)RB_ * M_ * 2);
  __bf16* ktT    = (__bf16*)carve((size_t)R_ * M_ * B_ * 2);
  float*  knorm  = (float*) carve((size_t)RB_ * 4);
  float*  alpha  = (float*) carve((size_t)B_ * R_ * 4);
  int*    least  = (int*)   carve((size_t)B_ * 4);
  float*  erase  = (float*) carve((size_t)N_ * 4);
  __bf16* Abf    = (__bf16*)carve((size_t)N_ * M_ * 2);
  float*  anorm  = (float*) carve((size_t)N_ * 4);
  float*  Ksim   = (float*) carve((size_t)RB_ * N_ * 4);
  __bf16* wrbf   = (__bf16*)carve((size_t)RB_ * N_ * 2);
  float*  Anew   = (float*) carve((size_t)N_ * M_ * 4);
  __bf16* AnewT  = (__bf16*)carve((size_t)M_ * N_ * 2);
  float*  reads  = (float*) carve((size_t)RB_ * M_ * 4);

  // 1) controller
  k_prep_catA<<<(B_ * KCAT_) / 256, 256, 0, stream>>>(x, read_prev, h, catA);
  k_prep_wcatT<<<(KCAT_ * G4_) / 256, 256, 0, stream>>>(W, Uw, wcatT);
  k_gemm_gates<<<64, 128, 0, stream>>>(catA, wcatT, gates);
  k_lstm<<<(B_ * U_) / 256, 256, 0, stream>>>(gates, bvec, c, h_new, out);

  // 2) heads / addressing scalars
  k_kt<<<(R_ * B_ * M_) / 256, 256, 0, stream>>>(h_new, Wk, bk, ktbf, ktT);
  k_knorm<<<RB_ / 256, 256, 0, stream>>>(ktbf, knorm);
  k_alpha<<<(B_ * R_) / 256, 256, 0, stream>>>(h_new, Wa, ba, alpha);
  k_argmin<<<B_, 256, 0, stream>>>(wu_prev, least);
  k_erase_init<<<N_ / 256, 256, 0, stream>>>(erase);
  k_erase_scatter<<<1, 256, 0, stream>>>(least, erase);
  k_aprep<<<N_, 32, 0, stream>>>(A, Abf, anorm);

  // 3) content addressing + softmax
  k_gemm_ksim<<<(64 * 512) / 4, 128, 0, stream>>>(ktbf, Abf, knorm, anorm, Ksim);
  k_softmax<<<RB_, 256, 0, stream>>>(Ksim, wrbf);

  // 4) sequential memory updates (tanh between writes forces 4 launches)
  for (int r = 0; r < R_; ++r) {
    k_update<<<N_ / 64, 128, 0, stream>>>(
        wr_prev + (size_t)r * B_ * N_, alpha, least, erase, A,
        ktT + (size_t)r * M_ * B_, Anew, AnewT,
        r, (r == 0) ? 1 : 0, (r == R_ - 1) ? 1 : 0);
  }

  // 5) reads (split-K=16, f32 atomics) + output assembly
  k_zero_reads<<<(RB_ * M_) / 256, 256, 0, stream>>>(reads);
  k_gemm_reads<<<1024 / 4, 128, 0, stream>>>(wrbf, AnewT, reads);
  k_out<<<(B_ * R_ * M_) / 256, 256, 0, stream>>>(reads, out);
}
